// MultiHeadAttention_44693429682618
// MI455X (gfx1250) — matrix operand order repaired and checked by reading, not compiled, not run on hardware
//
#include <hip/hip_runtime.h>
#include <hip/hip_bf16.h>
#include <math.h>

// ---------------- problem constants (from reference) ----------------
#define BB 2
#define TT 2048
#define CC 1024
#define HH 16
#define DD 64
#define MROWS (BB * TT)          // 4096
typedef __attribute__((ext_vector_type(16))) __bf16 v16bf;
typedef __attribute__((ext_vector_type(8)))  float  v8f;

// LDS byte offset of a __shared__ pointer (generic LDS addr[31:0] == wave-relative offset)
static __device__ __forceinline__ uint32_t lds_off(const void* p) {
    return (uint32_t)(uintptr_t)p;
}

// per-lane async copy of 16 bytes global -> LDS (tracked by ASYNCcnt)
static __device__ __forceinline__ void async_copy_b128(uint32_t lds_byte_off, const void* gptr) {
    unsigned long long ga = (unsigned long long)(uintptr_t)gptr;
    asm volatile("global_load_async_to_lds_b128 %0, %1, off"
                 :: "v"(lds_byte_off), "v"(ga) : "memory");
}

// ---------------- fp32 -> bf16 conversion ----------------
__global__ __launch_bounds__(256) void cvt_f32_to_bf16(const float* __restrict__ x,
                                                       __bf16* __restrict__ y, int n) {
    int i = (blockIdx.x * 256 + threadIdx.x) * 4;
    if (i + 3 < n) {
        float4 f = *(const float4*)(x + i);
        y[i + 0] = (__bf16)f.x;
        y[i + 1] = (__bf16)f.y;
        y[i + 2] = (__bf16)f.z;
        y[i + 3] = (__bf16)f.w;
    }
}

// ---------------- GEMM: Y[m,n] = sum_k X[m,k]*W[n,k] + bias[n] ----------------
// X: bf16 [M][K] row-major, W: fp32 [N][K] row-major.
// Block: 256 threads (8 waves), tile 128(M) x 64(N), K step 64.
template <typename OutT>
__global__ __launch_bounds__(256) void gemm_xWt_bf16(const __bf16* __restrict__ X,
                                                     const float* __restrict__ W,
                                                     const float* __restrict__ bias,
                                                     OutT* __restrict__ Y,
                                                     int Kd, int Nd) {
    __shared__ __bf16 Xs[128][72];   // 128 x 64 tile, padded stride (144B, 16B-aligned rows)
    __shared__ __bf16 Ws[64][72];    // 64 (n) x 64 (k) tile

    const int tid  = threadIdx.x;
    const int lane = tid & 31;
    const int wid  = tid >> 5;
    const int wm   = wid >> 1;        // 0..3  -> m offset wm*32
    const int wn   = wid & 1;         // 0..1  -> n offset wn*32
    const int half = lane >> 4;       // 0/1
    const int nl   = lane & 15;

    const int m0 = blockIdx.y * 128;
    const int n0 = blockIdx.x * 64;

    v8f acc[2][2] = {};

    for (int k0 = 0; k0 < Kd; k0 += 64) {
        // stage X tile: 128 rows x 64 bf16 = 1024 x b128 chunks
        for (int i = tid; i < 1024; i += 256) {
            int row = i >> 3, c4 = i & 7;
            *(uint4*)&Xs[row][c4 * 8] =
                *(const uint4*)(X + (size_t)(m0 + row) * Kd + k0 + c4 * 8);
        }
        // stage W tile (fp32 -> bf16): 64 rows x 64 floats, float4 loads
        for (int i = tid; i < 1024; i += 256) {
            int row = i >> 4, c4 = i & 15;
            float4 f = *(const float4*)(W + (size_t)(n0 + row) * Kd + k0 + c4 * 4);
            __bf16* d = &Ws[row][c4 * 4];
            d[0] = (__bf16)f.x; d[1] = (__bf16)f.y;
            d[2] = (__bf16)f.z; d[3] = (__bf16)f.w;
        }
        // prefetch next K tile into caches (global_prefetch_b8)
        if (k0 + 64 < Kd) {
            __builtin_prefetch(X + (size_t)(m0 + (tid & 127)) * Kd + k0 + 64 + (tid >> 7) * 32, 0, 0);
            __builtin_prefetch(W + (size_t)(n0 + (tid & 63)) * Kd + k0 + 64 + ((tid >> 6) & 3) * 16, 0, 0);
        }
        __syncthreads();

#pragma unroll
        for (int kk = 0; kk < 2; kk++) {   // two 32-wide K steps
            v16bf af[2], bfm[2];
#pragma unroll
            for (int im = 0; im < 2; im++)
#pragma unroll
                for (int v = 0; v < 8; v++)
#pragma unroll
                    for (int t = 0; t < 2; t++) {
                        int koff = kk * 32 + (v >> 2) * 16 + half * 8 + (v & 3) * 2 + t;
                        af[im][2 * v + t] = Xs[wm * 32 + im * 16 + nl][koff];
                    }
#pragma unroll
            for (int jn = 0; jn < 2; jn++)
#pragma unroll
                for (int v = 0; v < 8; v++)
#pragma unroll
                    for (int t = 0; t < 2; t++) {
                        int koff = kk * 32 + (v >> 2) * 16 + half * 8 + (v & 3) * 2 + t;
                        bfm[jn][2 * v + t] = Ws[wn * 32 + jn * 16 + nl][koff];
                    }
#pragma unroll
            for (int im = 0; im < 2; im++)
#pragma unroll
                for (int jn = 0; jn < 2; jn++)
                    acc[im][jn] = __builtin_amdgcn_wmma_f32_16x16x32_bf16(
                        false, af[im], false, bfm[jn], (short)0, acc[im][jn], false, false);
        }
        __syncthreads();
    }

    // epilogue: C layout -> lane holds rows half*8+r at column nl
#pragma unroll
    for (int im = 0; im < 2; im++)
#pragma unroll
        for (int jn = 0; jn < 2; jn++)
#pragma unroll
            for (int r = 0; r < 8; r++) {
                int row = m0 + wm * 32 + im * 16 + half * 8 + r;
                int col = n0 + wn * 32 + jn * 16 + nl;
                float val = acc[im][jn][r] + bias[col];
                Y[(size_t)row * Nd + col] = (OutT)val;
            }
}

// ---------------- RoPE (in place on bf16 q or k) ----------------
__global__ __launch_bounds__(256) void rope_bf16(__bf16* __restrict__ a) {
    int g   = blockIdx.x * 256 + threadIdx.x;     // B*T*H*32 threads
    int j   = g & 31;
    int rh  = g >> 5;
    int h   = rh & (HH - 1);
    int row = rh >> 4;                            // HH == 16
    int tpos = row & (TT - 1);

    float theta = __powf(10000.0f, -(2.0f * (float)j) / (float)DD);
    float ang = (float)tpos * theta;
    float s, c;
    __sincosf(ang, &s, &c);

    __bf16* p = a + (size_t)row * CC + h * DD;
    float x1 = (float)p[2 * j];
    float x2 = (float)p[2 * j + 1];
    float o0 = x1 * c - x2 * s;
    float o1 = x1 * s + x2 * c;
    p[j]      = (__bf16)o0;
    p[j + 32] = (__bf16)o1;
}

// ---------------- Flash attention: one wave per 16 q-rows per (b,h) ----------------
// K/V blocks (32 x 64 bf16) staged via async global->LDS DMA, double buffered.
__global__ __launch_bounds__(32) void attn_flash_bf16(const __bf16* __restrict__ q,
                                                      const __bf16* __restrict__ k,
                                                      const __bf16* __restrict__ v,
                                                      __bf16* __restrict__ o) {
    __shared__ __bf16 Ks[2][32][72];   // [buf][key][c], 16B-aligned rows
    __shared__ __bf16 Vs[2][32][72];   // [buf][key][d]
    __shared__ __bf16 Pl[16][32];

    const int lane = threadIdx.x;
    const int half = lane >> 4;
    const int nl   = lane & 15;
    const int qb   = blockIdx.x * 16;
    const int h    = blockIdx.y;
    const int b    = blockIdx.z;
    const size_t hoff = (size_t)h * DD;

    // load Q fragments (A layout, 16 x 64 => 2 k-steps of 32)
    v16bf qa[2];
#pragma unroll
    for (int kk = 0; kk < 2; kk++)
#pragma unroll
        for (int vv = 0; vv < 8; vv++)
#pragma unroll
            for (int t = 0; t < 2; t++) {
                int koff = kk * 32 + (vv >> 2) * 16 + half * 8 + (vv & 3) * 2 + t;
                qa[kk][2 * vv + t] = q[((size_t)(b * TT + qb + nl)) * CC + hoff + koff];
            }

    v8f oacc[4] = {};
    float mrow[8], lrow[8];
#pragma unroll
    for (int r = 0; r < 8; r++) { mrow[r] = -3.0e38f; lrow[r] = 0.0f; }

    const float scale = 0.125f;  // 1/sqrt(64)
    const int nkb = qb / 32 + 1; // 32-key blocks (causal)

    // issue async stage of a 32x64 K block and V block into buffer `buf`
    auto stage = [&](int buf, int kb) {
        // 256 b128 chunks per array, 32 lanes -> 8 instructions each
#pragma unroll
        for (int jj = 0; jj < 8; jj++) {
            int idx = jj * 32 + lane;
            int r = idx >> 3, c4 = idx & 7;
            async_copy_b128(lds_off(&Ks[buf][r][c4 * 8]),
                            k + (size_t)(b * TT + kb + r) * CC + hoff + c4 * 8);
        }
#pragma unroll
        for (int jj = 0; jj < 8; jj++) {
            int idx = jj * 32 + lane;
            int r = idx >> 3, c4 = idx & 7;
            async_copy_b128(lds_off(&Vs[buf][r][c4 * 8]),
                            v + (size_t)(b * TT + kb + r) * CC + hoff + c4 * 8);
        }
    };

    stage(0, 0);   // prime buffer 0

    for (int kb32 = 0; kb32 < nkb; kb32++) {
        const int kb  = kb32 * 32;
        const int buf = kb32 & 1;

        // overlap: issue DMA for next block, then wait only for current block (16 loads left in flight)
        if (kb32 + 1 < nkb) {
            stage(buf ^ 1, kb + 32);
            asm volatile("s_wait_asynccnt 0x10" ::: "memory");
        } else {
            asm volatile("s_wait_asynccnt 0x0" ::: "memory");
        }
        __syncthreads();

        v8f s0 = {}, s1 = {};
        // scores: S = Q * K^T  (B(c,key) = K[key][c]) from LDS
#pragma unroll
        for (int kk = 0; kk < 2; kk++) {
            v16bf kb0, kb1;
#pragma unroll
            for (int vv = 0; vv < 8; vv++)
#pragma unroll
                for (int t = 0; t < 2; t++) {
                    int c = kk * 32 + (vv >> 2) * 16 + half * 8 + (vv & 3) * 2 + t;
                    kb0[2 * vv + t] = Ks[buf][nl][c];
                    kb1[2 * vv + t] = Ks[buf][16 + nl][c];
                }
            s0 = __builtin_amdgcn_wmma_f32_16x16x32_bf16(false, qa[kk], false, kb0,
                                                         (short)0, s0, false, false);
            s1 = __builtin_amdgcn_wmma_f32_16x16x32_bf16(false, qa[kk], false, kb1,
                                                         (short)0, s1, false, false);
        }

        // mask + online softmax (row M = half*8 + r, col = nl within tile)
        float alpha[8];
#pragma unroll
        for (int r = 0; r < 8; r++) {
            int qrow = qb + half * 8 + r;
            float f0 = s0[r] * scale;
            float f1 = s1[r] * scale;
            if (kb + nl > qrow)      f0 = -1.0e30f;
            if (kb + 16 + nl > qrow) f1 = -1.0e30f;
            float mx = fmaxf(f0, f1);
#pragma unroll
            for (int msk = 1; msk <= 8; msk <<= 1)
                mx = fmaxf(mx, __shfl_xor(mx, msk, 32));
            float mnew = fmaxf(mrow[r], mx);
            float a = __expf(mrow[r] - mnew);
            float p0 = __expf(f0 - mnew);
            float p1 = __expf(f1 - mnew);
            float rs = p0 + p1;
#pragma unroll
            for (int msk = 1; msk <= 8; msk <<= 1)
                rs += __shfl_xor(rs, msk, 32);
            lrow[r] = lrow[r] * a + rs;
            mrow[r] = mnew;
            alpha[r] = a;
            s0[r] = p0;
            s1[r] = p1;
        }
#pragma unroll
        for (int nt = 0; nt < 4; nt++)
#pragma unroll
            for (int r = 0; r < 8; r++)
                oacc[nt][r] *= alpha[r];

        // repack P: C layout -> A layout via LDS
#pragma unroll
        for (int r = 0; r < 8; r++) {
            Pl[half * 8 + r][nl]      = (__bf16)s0[r];
            Pl[half * 8 + r][16 + nl] = (__bf16)s1[r];
        }
        __syncthreads();
        v16bf pa;
#pragma unroll
        for (int vv = 0; vv < 8; vv++)
#pragma unroll
            for (int t = 0; t < 2; t++) {
                int koff = (vv >> 2) * 16 + half * 8 + (vv & 3) * 2 + t;
                pa[2 * vv + t] = Pl[nl][koff];
            }
        __syncthreads();

        // O += P * V  (B(key,d) = V[kb+key][d]) from LDS, D=64 -> 4 N tiles
#pragma unroll
        for (int nt = 0; nt < 4; nt++) {
            v16bf vb;
#pragma unroll
            for (int vv = 0; vv < 8; vv++)
#pragma unroll
                for (int t = 0; t < 2; t++) {
                    int key = (vv >> 2) * 16 + half * 8 + (vv & 3) * 2 + t;
                    vb[2 * vv + t] = Vs[buf][key][nt * 16 + nl];
                }
            oacc[nt] = __builtin_amdgcn_wmma_f32_16x16x32_bf16(false, pa, false, vb,
                                                               (short)0, oacc[nt], false, false);
        }
    }

    // epilogue: normalize by l and store bf16 context
#pragma unroll
    for (int nt = 0; nt < 4; nt++)
#pragma unroll
        for (int r = 0; r < 8; r++) {
            int row = b * TT + qb + half * 8 + r;
            float val = oacc[nt][r] / lrow[r];
            o[(size_t)row * CC + hoff + nt * 16 + nl] = (__bf16)val;
        }
}

// ---------------- host-side orchestration ----------------
extern "C" void kernel_launch(void* const* d_in, const int* in_sizes, int n_in,
                              void* d_out, int out_size, void* d_ws, size_t ws_size,
                              hipStream_t stream) {
    const float* x  = (const float*)d_in[0];
    const float* Wq = (const float*)d_in[1];
    const float* bq = (const float*)d_in[2];
    const float* Wk = (const float*)d_in[3];
    const float* bk = (const float*)d_in[4];
    const float* Wv = (const float*)d_in[5];
    const float* bv = (const float*)d_in[6];
    const float* Wo = (const float*)d_in[7];
    const float* bo = (const float*)d_in[8];
    float* out = (float*)d_out;

    const size_t SZ = (size_t)MROWS * CC;   // 4M elements
    __bf16* xb   = (__bf16*)d_ws;
    __bf16* qbuf = xb + SZ;
    __bf16* kbuf = qbuf + SZ;
    __bf16* vbuf = kbuf + SZ;
    __bf16* abuf = vbuf + SZ;               // attention context

    // 1) x -> bf16
    cvt_f32_to_bf16<<<(int)(SZ / 1024), 256, 0, stream>>>(x, xb, (int)SZ);

    // 2) QKV projections (bf16 out)
    dim3 ggrid(CC / 64, MROWS / 128);
    gemm_xWt_bf16<__bf16><<<ggrid, 256, 0, stream>>>(xb, Wq, bq, qbuf, CC, CC);
    gemm_xWt_bf16<__bf16><<<ggrid, 256, 0, stream>>>(xb, Wk, bk, kbuf, CC, CC);
    gemm_xWt_bf16<__bf16><<<ggrid, 256, 0, stream>>>(xb, Wv, bv, vbuf, CC, CC);

    // 3) RoPE on q and k (in place)
    int rope_threads = MROWS * HH * 32;
    rope_bf16<<<rope_threads / 256, 256, 0, stream>>>(qbuf);
    rope_bf16<<<rope_threads / 256, 256, 0, stream>>>(kbuf);

    // 4) causal flash attention (async-LDS double-buffered K/V)
    dim3 agrid(TT / 16, HH, BB);
    attn_flash_bf16<<<agrid, 32, 0, stream>>>(qbuf, kbuf, vbuf, abuf);

    // 5) output projection (fp32 out)
    gemm_xWt_bf16<float><<<ggrid, 256, 0, stream>>>(abuf, Wo, bo, out, CC, CC);
}
